// AudioModelM3_74783970558077
// MI455X (gfx1250) — compile-verified, matrix operand order checked by reading, not run on hardware
//
#include <hip/hip_runtime.h>

typedef __bf16 bf16_t;
typedef __attribute__((ext_vector_type(16))) __bf16 v16bf;
typedef __attribute__((ext_vector_type(8)))  __bf16 v8bf;
typedef __attribute__((ext_vector_type(8)))  float  v8f;
typedef __attribute__((ext_vector_type(4)))  unsigned v4u;
typedef __attribute__((ext_vector_type(8)))  int    v8i_t;
typedef __attribute__((ext_vector_type(4)))  int    v4i_t;

#define D_MODEL   1024
#define D_INNER   2048
#define D_STATE   128
#define NHEADS    32
#define HEADDIM   64
#define D_CONV    4
#define D_CONV_CH 2304
#define D_IN_PROJ 4384
#define BATCH     8
#define SEQ       1024
#define NTOK      (BATCH * SEQ)   // 8192
#define CHUNKQ    64
#define NCHUNK    (SEQ / CHUNKQ)  // 16

// ---------- helpers ----------------------------------------------------------

__device__ __forceinline__ bf16_t f2bf(float f) {
  unsigned u = __builtin_bit_cast(unsigned, f);
  unsigned r = (u + 0x7FFFu + ((u >> 16) & 1u)) >> 16;
  unsigned short s = (unsigned short)r;
  return __builtin_bit_cast(bf16_t, s);
}

__device__ __forceinline__ v8f wmma_bf16(v16bf a, v16bf b, v8f c) {
  // D = A(16x32) * B(32x16) + C, f32 accumulate (wave32 CDNA5 WMMA)
  return __builtin_amdgcn_wmma_f32_16x16x32_bf16(false, a, false, b, (short)0, c,
                                                 false, false);
}

// Fragment loader: LDS tile stored [row][k] (row-major over K), stride RS elems.
// RS*2 bytes and the K-half offsets are 16B-multiples -> two ds_load_b128 per frag.
template <int RS>
__device__ __forceinline__ v16bf frag_rk(const bf16_t* base, int lane) {
  const bf16_t* p = base + (lane & 15) * RS + ((lane >> 4) << 3);
  v8bf lo = *(const v8bf*)p;
  v8bf hi = *(const v8bf*)(p + 16);
  v16bf f;
#pragma unroll
  for (int j = 0; j < 8; ++j) { f[j] = lo[j]; f[8 + j] = hi[j]; }
  return f;
}

// ---------- Tensor Data Mover: 2-D bf16 tile -> LDS (D# per CDNA5 ISA ch.8) --
// Loads tile_d1 rows x 32 elems (64B) per row, padding LDS by 16B per row so the
// destination is a [rows][40] bf16 array (ds_load_b128-aligned). OOB rows -> 0.
__device__ __forceinline__ void tdm_load_tile(unsigned lds_byte_off,
                                              const bf16_t* gaddr,
                                              unsigned tensor_d0,   // row len (elems)
                                              unsigned tensor_d1,   // valid rows
                                              unsigned tile_d1,     // rows to move
                                              unsigned stride_d0) { // row stride (elems)
  unsigned long long ga = (unsigned long long)(size_t)gaddr;
  v4u g0;
  g0[0] = 1u;                                   // count=1 (valid user descriptor)
  g0[1] = lds_byte_off;                         // lds_addr
  g0[2] = (unsigned)(ga & 0xFFFFFFFFu);         // global_addr[31:0]
  g0[3] = (unsigned)((ga >> 32) & 0x01FFFFFFu)  // global_addr[56:32]
          | (2u << 30);                         // type=2 ("image")
  v8i_t g1;
  g1[0] = (int)((1u << 16)      // data_size = 2 bytes
                | (1u << 20)    // pad_enable
                | (3u << 22)    // pad_interval: 16 DWORDs (= 32 bf16 row)
                | (3u << 25));  // pad_amount: 4 DWORDs (= 8 bf16 pad)
  g1[1] = (int)((tensor_d0 & 0xFFFFu) << 16);                     // tensor_dim0 lo
  g1[2] = (int)((tensor_d0 >> 16) | ((tensor_d1 & 0xFFFFu) << 16));
  g1[3] = (int)((tensor_d1 >> 16) | (32u << 16));                 // tile_dim0 = 32
  g1[4] = (int)(tile_d1 & 0xFFFFu);                               // tile_dim1
  g1[5] = (int)stride_d0;                                         // dim0_stride lo
  g1[6] = 0;
  g1[7] = 0;
  v4i_t z4 = {0, 0, 0, 0};
#if defined(__clang_major__) && (__clang_major__ >= 23)
  v8i_t z8 = {0, 0, 0, 0, 0, 0, 0, 0};
  __builtin_amdgcn_tensor_load_to_lds(g0, g1, z4, z4, z8, 0);
#else
  __builtin_amdgcn_tensor_load_to_lds(g0, g1, z4, z4, 0);
#endif
}

// ---------- f32 -> bf16 staging ---------------------------------------------
__global__ void __launch_bounds__(256) cvt_bf16_k(const float* __restrict__ src,
                                                  bf16_t* __restrict__ dst, long n) {
  long i = (long)blockIdx.x * 256 + threadIdx.x;
  if (i < n) dst[i] = f2bf(src[i]);
}

// ---------- GEMM: C[M,N] = A[M,K] * B[N,K]^T  (bf16 in, TDM-staged, WMMA) ---
// EPI: 0 = none, 1 = SELU
template <int EPI>
__global__ void __launch_bounds__(256) gemm_abt(const bf16_t* __restrict__ A,
                                                const bf16_t* __restrict__ B,
                                                float* __restrict__ C,
                                                int M, int N, int K) {
  __shared__ alignas(16) bf16_t As[128][40];
  __shared__ alignas(16) bf16_t Bs[128][40];
  const int tid = threadIdx.x, lane = tid & 31, wave = tid >> 5;
  const int bm = blockIdx.y * 128, bn = blockIdx.x * 128;
  const int wm = (wave >> 2) * 64;  // 0 / 64
  const int wn = (wave & 3) * 32;   // 0,32,64,96
  const unsigned ldsA = (unsigned)(size_t)(void*)&As[0][0];
  const unsigned ldsB = (unsigned)(size_t)(void*)&Bs[0][0];
  const unsigned bRows = (unsigned)((N - bn < 128) ? (N - bn) : 128);
  v8f acc[4][2] = {};
  for (int k0 = 0; k0 < K; k0 += 32) {
    if (wave == 0) {  // TDM issue is wave-level; one wave drives both tiles
      tdm_load_tile(ldsA, A + (size_t)bm * K + k0, (unsigned)K, 128u, 128u, (unsigned)K);
      tdm_load_tile(ldsB, B + (size_t)bn * K + k0, (unsigned)K, bRows, 128u, (unsigned)K);
      __builtin_amdgcn_s_wait_tensorcnt(0);
    }
    if (k0 + 32 < K)  // gfx1250 global_prefetch_b8 of the next A tile
      __builtin_prefetch(&A[(size_t)(bm + (tid >> 1)) * K + k0 + 32 + (tid & 1) * 16], 0, 0);
    __syncthreads();
    v16bf b0 = frag_rk<40>(&Bs[wn][0], lane);
    v16bf b1 = frag_rk<40>(&Bs[wn + 16][0], lane);
#pragma unroll
    for (int i = 0; i < 4; ++i) {
      v16bf af = frag_rk<40>(&As[wm + 16 * i][0], lane);
      acc[i][0] = wmma_bf16(af, b0, acc[i][0]);
      acc[i][1] = wmma_bf16(af, b1, acc[i][1]);
    }
    __syncthreads();
  }
  const int cl = lane & 15, rb = (lane >> 4) << 3;
#pragma unroll
  for (int i = 0; i < 4; ++i) {
#pragma unroll
    for (int f = 0; f < 2; ++f) {
      int col = bn + wn + 16 * f + cl;
      if (col >= N) continue;
#pragma unroll
      for (int r = 0; r < 8; ++r) {
        int row = bm + wm + 16 * i + rb + r;
        float v = acc[i][f][r];
        if (EPI == 1) {
          const float kA = 1.6732632423543772f, kS = 1.0507009873554805f;
          v = (v > 0.f) ? kS * v : kS * kA * (__expf(v) - 1.f);
        }
        C[(size_t)row * N + col] = v;
      }
    }
  }
}

// ---------- conv1d (depthwise, causal, k=4) + SiLU ---------------------------
__global__ void __launch_bounds__(256) conv_silu_k(const float* __restrict__ zx,
                                                   const float* __restrict__ cw,
                                                   const float* __restrict__ cb,
                                                   float* __restrict__ xact) {
  long idx = (long)blockIdx.x * 256 + threadIdx.x;
  if (idx >= (long)NTOK * D_CONV_CH) return;
  int ch = (int)(idx % D_CONV_CH);
  long tok = idx / D_CONV_CH;
  int t = (int)(tok & (SEQ - 1));
  float acc = cb[ch];
#pragma unroll
  for (int j = 0; j < D_CONV; ++j) {
    int tt = t - (D_CONV - 1) + j;
    if (tt >= 0)
      acc += cw[ch * D_CONV + j] *
             zx[(size_t)(tok - (D_CONV - 1) + j) * D_IN_PROJ + D_INNER + ch];
  }
  xact[(size_t)tok * D_CONV_CH + ch] = acc / (1.f + __expf(-acc));
}

// ---------- dt = softplus(dt + bias) ----------------------------------------
__global__ void __launch_bounds__(256) dt_softplus_k(const float* __restrict__ zx,
                                                     const float* __restrict__ dt_bias,
                                                     float* __restrict__ dtb) {
  int idx = blockIdx.x * 256 + threadIdx.x;
  if (idx >= NTOK * NHEADS) return;
  int h = idx & (NHEADS - 1);
  int row = idx >> 5;
  float v = zx[(size_t)row * D_IN_PROJ + (D_INNER + D_CONV_CH) + h] + dt_bias[h];
  dtb[idx] = (v > 20.f) ? v : log1pf(__expf(v));
}

// ---------- SSD intra-chunk: Yd + per-chunk states (all WMMA) ---------------
__global__ void __launch_bounds__(256) ssd_intra(const float* __restrict__ xact,
                                                 const float* __restrict__ dtb,
                                                 const float* __restrict__ A_log,
                                                 float* __restrict__ ybuf,
                                                 float* __restrict__ states,
                                                 float* __restrict__ acsbuf) {
  __shared__ float sAcs[64];
  __shared__ alignas(16) bf16_t Bq[64][136];   // B tile, q-major
  __shared__ alignas(16) bf16_t Ph[128 * 72];  // phase2: C tile [64][136]; phase3: B^T [128][72]
  __shared__ alignas(16) bf16_t XsN[64][72];   // X^T (p-major), no decay
  __shared__ alignas(16) bf16_t XsT[64][72];   // X^T (p-major), * decay_states
  __shared__ alignas(16) bf16_t Gs[64][72];    // masked/decayed CB^T
  bf16_t(*Cs)[136] = (bf16_t(*)[136])Ph;
  bf16_t(*BsT)[72] = (bf16_t(*)[72])Ph;
  const int h = blockIdx.x, c = blockIdx.y, b = blockIdx.z;
  const int tid = threadIdx.x, lane = tid & 31, wave = tid >> 5;
  const int row0 = (b * NCHUNK + c) * CHUNKQ;
  const size_t hcb = (size_t)(b * NCHUNK + c) * NHEADS + h;
  const float Ah = -__expf(A_log[h]);
  if (tid < 64) sAcs[tid] = dtb[(size_t)(row0 + tid) * NHEADS + h] * Ah;
  __syncthreads();
  for (int o = 1; o < 64; o <<= 1) {  // inclusive Hillis-Steele scan
    float v = (tid < 64 && tid >= o) ? sAcs[tid - o] : 0.f;
    __syncthreads();
    if (tid < 64) sAcs[tid] += v;
    __syncthreads();
  }
  if (tid < 64) acsbuf[hcb * 64 + tid] = sAcs[tid];
  const float acsLast = sAcs[63];
  for (int i = tid; i < 64 * 128; i += 256) {
    int q = i >> 7, n = i & 127;
    size_t r = (size_t)(row0 + q) * D_CONV_CH;
    Bq[q][n] = f2bf(xact[r + D_INNER + n]);
    Cs[q][n] = f2bf(xact[r + D_INNER + D_STATE + n]);
  }
  for (int i = tid; i < 64 * 64; i += 256) {
    int q = i >> 6, p = i & 63;
    float dtq = dtb[(size_t)(row0 + q) * NHEADS + h];
    float xv = xact[(size_t)(row0 + q) * D_CONV_CH + h * HEADDIM + p] * dtq;
    XsN[p][q] = f2bf(xv);
    XsT[p][q] = f2bf(xv * __expf(acsLast - sAcs[q]));  // decay_states
  }
  __syncthreads();
  // Phase 2: G = C * B^T (64x64, K=128) -> mask + segsum decay -> Gs (bf16)
  {
    const int gm = (wave & 3) * 16, gn = (wave >> 2) * 32;
    v8f g0 = {}, g1 = {};
#pragma unroll
    for (int kk = 0; kk < 128; kk += 32) {
      v16bf a  = frag_rk<136>(&Cs[gm][kk], lane);
      v16bf b0 = frag_rk<136>(&Bq[gn][kk], lane);
      v16bf b1 = frag_rk<136>(&Bq[gn + 16][kk], lane);
      g0 = wmma_bf16(a, b0, g0);
      g1 = wmma_bf16(a, b1, g1);
    }
    const int cl = lane & 15, rb = (lane >> 4) << 3;
#pragma unroll
    for (int r = 0; r < 8; ++r) {
      int i = gm + rb + r;
      float ai = sAcs[i];
      int j0 = gn + cl, j1 = gn + cl + 16;
      Gs[i][j0] = f2bf(i >= j0 ? g0[r] * __expf(ai - sAcs[j0]) : 0.f);
      Gs[i][j1] = f2bf(i >= j1 ? g1[r] * __expf(ai - sAcs[j1]) : 0.f);
    }
  }
  __syncthreads();
  // Phase 3a: transpose B into the retired C region; Yd = Gs @ X (64x64, K=64)
  for (int i = tid; i < 64 * 128; i += 256) {
    int n = i >> 6, q = i & 63;
    BsT[n][q] = Bq[q][n];
  }
  {
    const int gm = (wave & 3) * 16, gn = (wave >> 2) * 32;
    v8f y0 = {}, y1 = {};
#pragma unroll
    for (int kk = 0; kk < 64; kk += 32) {
      v16bf a  = frag_rk<72>(&Gs[gm][kk], lane);
      v16bf b0 = frag_rk<72>(&XsN[gn][kk], lane);
      v16bf b1 = frag_rk<72>(&XsN[gn + 16][kk], lane);
      y0 = wmma_bf16(a, b0, y0);
      y1 = wmma_bf16(a, b1, y1);
    }
    const int cl = lane & 15, rb = (lane >> 4) << 3;
#pragma unroll
    for (int r = 0; r < 8; ++r) {
      size_t row = (size_t)(row0 + gm + rb + r);
      ybuf[row * D_INNER + h * HEADDIM + gn + cl]      = y0[r];
      ybuf[row * D_INNER + h * HEADDIM + gn + cl + 16] = y1[r];
    }
  }
  __syncthreads();
  // Phase 3b: states = (decay*X)^T @ B  (64x128, K=64)
  {
    const int sm = (wave & 1) * 32, sn = (wave >> 1) * 32;
    v8f s00 = {}, s01 = {}, s10 = {}, s11 = {};
#pragma unroll
    for (int kk = 0; kk < 64; kk += 32) {
      v16bf a0 = frag_rk<72>(&XsT[sm][kk], lane);
      v16bf a1 = frag_rk<72>(&XsT[sm + 16][kk], lane);
      v16bf b0 = frag_rk<72>(&BsT[sn][kk], lane);
      v16bf b1 = frag_rk<72>(&BsT[sn + 16][kk], lane);
      s00 = wmma_bf16(a0, b0, s00); s01 = wmma_bf16(a0, b1, s01);
      s10 = wmma_bf16(a1, b0, s10); s11 = wmma_bf16(a1, b1, s11);
    }
    const int cl = lane & 15, rb = (lane >> 4) << 3;
    float* sp = states + hcb * (64 * 128);
#pragma unroll
    for (int r = 0; r < 8; ++r) {
      sp[(size_t)(sm + rb + r) * 128 + sn + cl]           = s00[r];
      sp[(size_t)(sm + rb + r) * 128 + sn + cl + 16]      = s01[r];
      sp[(size_t)(sm + 16 + rb + r) * 128 + sn + cl]      = s10[r];
      sp[(size_t)(sm + 16 + rb + r) * 128 + sn + cl + 16] = s11[r];
    }
  }
}

// ---------- sequential inter-chunk scan: states -> exclusive prefix ---------
__global__ void __launch_bounds__(256) chunk_scan(float* __restrict__ states,
                                                  const float* __restrict__ acsbuf) {
  const int h = blockIdx.x, b = blockIdx.y;
  float acc[32];
#pragma unroll
  for (int j = 0; j < 32; ++j) acc[j] = 0.f;
  for (int c = 0; c < NCHUNK; ++c) {
    size_t hcb = (size_t)(b * NCHUNK + c) * NHEADS + h;
    float cdec = __expf(acsbuf[hcb * 64 + 63]);
    size_t base = hcb * (64 * 128);
#pragma unroll
    for (int j = 0; j < 32; ++j) {
      size_t e = base + threadIdx.x + 256 * j;
      float s = states[e];
      states[e] = acc[j];  // prev-state (before this chunk)
      acc[j] = acc[j] * cdec + s;
    }
  }
}

// ---------- SSD inter-chunk: Yo = (e^Acs * C) @ prev^T  + D skip ------------
__global__ void __launch_bounds__(256) ssd_inter(const float* __restrict__ xact,
                                                 const float* __restrict__ prev,
                                                 const float* __restrict__ acsbuf,
                                                 const float* __restrict__ Dv,
                                                 float* __restrict__ ybuf) {
  __shared__ float sAcs[64];
  __shared__ alignas(16) bf16_t Ce[64][136];
  __shared__ alignas(16) bf16_t Pp[64][136];  // prev, p-major (natural B-frag layout)
  const int h = blockIdx.x, c = blockIdx.y, b = blockIdx.z;
  const int tid = threadIdx.x, lane = tid & 31, wave = tid >> 5;
  const int row0 = (b * NCHUNK + c) * CHUNKQ;
  const size_t hcb = (size_t)(b * NCHUNK + c) * NHEADS + h;
  if (tid < 64) sAcs[tid] = acsbuf[hcb * 64 + tid];
  __syncthreads();
  const float* pp = prev + hcb * (64 * 128);
  for (int i = tid; i < 64 * 128; i += 256) {
    int q = i >> 7, n = i & 127;
    Ce[q][n] = f2bf(xact[(size_t)(row0 + q) * D_CONV_CH + D_INNER + D_STATE + n] *
                    __expf(sAcs[q]));
    int p2 = i >> 7, n2 = i & 127;
    Pp[p2][n2] = f2bf(pp[(size_t)p2 * 128 + n2]);
  }
  __syncthreads();
  const int gm = (wave & 3) * 16, gn = (wave >> 2) * 32;
  v8f y0 = {}, y1 = {};
#pragma unroll
  for (int kk = 0; kk < 128; kk += 32) {
    v16bf a  = frag_rk<136>(&Ce[gm][kk], lane);
    v16bf b0 = frag_rk<136>(&Pp[gn][kk], lane);
    v16bf b1 = frag_rk<136>(&Pp[gn + 16][kk], lane);
    y0 = wmma_bf16(a, b0, y0);
    y1 = wmma_bf16(a, b1, y1);
  }
  const float Dh = Dv[h];
  const int cl = lane & 15, rb = (lane >> 4) << 3;
#pragma unroll
  for (int r = 0; r < 8; ++r) {
    size_t row = (size_t)(row0 + gm + rb + r);
    int p0 = gn + cl, p1 = gn + cl + 16;
    float x0 = xact[row * D_CONV_CH + h * HEADDIM + p0];
    float x1 = xact[row * D_CONV_CH + h * HEADDIM + p1];
    ybuf[row * D_INNER + h * HEADDIM + p0] += y0[r] + Dh * x0;
    ybuf[row * D_INNER + h * HEADDIM + p1] += y1[r] + Dh * x1;
  }
}

// ---------- gated RMSNorm -> bf16 GEMM operand ------------------------------
__global__ void __launch_bounds__(256) gated_rmsnorm(const float* __restrict__ y,
                                                     const float* __restrict__ zx,
                                                     const float* __restrict__ nw,
                                                     bf16_t* __restrict__ ybf) {
  __shared__ float red[256];
  const int row = blockIdx.x, tid = threadIdx.x;
  float t[8];
  float ss = 0.f;
#pragma unroll
  for (int j0 = 0; j0 < 8; ++j0) {
    int j = tid + 256 * j0;
    float z = zx[(size_t)row * D_IN_PROJ + j];
    float g = z / (1.f + __expf(-z));
    float v = y[(size_t)row * D_INNER + j] * g;
    t[j0] = v;
    ss += v * v;
  }
  red[tid] = ss;
  __syncthreads();
  for (int s = 128; s > 0; s >>= 1) {
    if (tid < s) red[tid] += red[tid + s];
    __syncthreads();
  }
  float scale = rsqrtf(red[0] / (float)D_INNER + 1e-5f);
#pragma unroll
  for (int j0 = 0; j0 < 8; ++j0) {
    int j = tid + 256 * j0;
    ybf[(size_t)row * D_INNER + j] = f2bf(t[j0] * scale * nw[j]);
  }
}

// ---------- mean pool over sequence -----------------------------------------
__global__ void __launch_bounds__(256) meanpool_k(const float* __restrict__ hbuf,
                                                  float* __restrict__ pooled) {
  int idx = blockIdx.x * 256 + threadIdx.x;
  if (idx >= BATCH * D_MODEL) return;
  int d = idx & (D_MODEL - 1);
  int b = idx >> 10;
  float s = 0.f;
  for (int l = 0; l < SEQ; ++l) s += hbuf[(size_t)(b * SEQ + l) * D_MODEL + d];
  pooled[idx] = s * (1.f / SEQ);
}

// ---------- classifier heads -------------------------------------------------
__global__ void __launch_bounds__(256) heads_k(const float* __restrict__ pooled,
                                               const float* __restrict__ we,
                                               const float* __restrict__ be,
                                               const float* __restrict__ wsn,
                                               const float* __restrict__ bsn,
                                               float* __restrict__ out) {
  __shared__ float red[256];
  const int o = blockIdx.x;  // 0..79
  const int b = o / 10, j = o % 10;
  const float* w = (j < 7) ? (we + (size_t)j * D_MODEL) : (wsn + (size_t)(j - 7) * D_MODEL);
  float bias = (j < 7) ? be[j] : bsn[j - 7];
  const int tid = threadIdx.x;
  float s = 0.f;
  for (int i = tid; i < D_MODEL; i += 256) s += pooled[(size_t)b * D_MODEL + i] * w[i];
  red[tid] = s;
  __syncthreads();
  for (int st = 128; st > 0; st >>= 1) {
    if (tid < st) red[tid] += red[tid + st];
    __syncthreads();
  }
  if (tid == 0) out[o] = red[0] + bias;
}

// ---------- host orchestration ----------------------------------------------
extern "C" void kernel_launch(void* const* d_in, const int* in_sizes, int n_in,
                              void* d_out, int out_size, void* d_ws, size_t ws_size,
                              hipStream_t stream) {
  (void)in_sizes; (void)n_in; (void)out_size; (void)ws_size;
  const float* x     = (const float*)d_in[0];
  const float* w_emo = (const float*)d_in[17];
  const float* b_emo = (const float*)d_in[18];
  const float* w_sen = (const float*)d_in[19];
  const float* b_sen = (const float*)d_in[20];

  char* ws = (char*)d_ws;
  size_t off = 0;
  auto alloc = [&](size_t bytes) -> char* {
    char* p = ws + off;
    off = (off + bytes + 255) & ~(size_t)255;
    return p;
  };
  float*  zx     = (float*)alloc((size_t)NTOK * D_IN_PROJ * 4);
  float*  xact   = (float*)alloc((size_t)NTOK * D_CONV_CH * 4);
  float*  dtb    = (float*)alloc((size_t)NTOK * NHEADS * 4);
  float*  acs    = (float*)alloc((size_t)BATCH * NCHUNK * NHEADS * 64 * 4);
  float*  states = (float*)alloc((size_t)BATCH * NCHUNK * NHEADS * 64 * 128 * 4);
  float*  ybuf   = (float*)alloc((size_t)NTOK * D_INNER * 4);
  float*  h1     = (float*)alloc((size_t)NTOK * D_MODEL * 4);
  float*  pooled = (float*)alloc((size_t)BATCH * D_MODEL * 4);
  bf16_t* abuf   = (bf16_t*)alloc((size_t)NTOK * D_MODEL * 2);
  bf16_t* ybf    = (bf16_t*)alloc((size_t)NTOK * D_INNER * 2);
  bf16_t* wbufA  = (bf16_t*)alloc((size_t)D_IN_PROJ * D_MODEL * 2);
  bf16_t* wbufB  = (bf16_t*)alloc((size_t)D_MODEL * D_INNER * 2);
  float*  h2     = zx;  // zx is dead by the time block-2's out_proj writes

  const float* cur = x;
  float* outs[2] = {h1, h2};
  for (int blk = 0; blk < 2; ++blk) {
    const float* in_w    = (const float*)d_in[1 + 8 * blk + 0];
    const float* conv_w  = (const float*)d_in[1 + 8 * blk + 1];
    const float* conv_b  = (const float*)d_in[1 + 8 * blk + 2];
    const float* dt_bias = (const float*)d_in[1 + 8 * blk + 3];
    const float* A_log   = (const float*)d_in[1 + 8 * blk + 4];
    const float* Dw      = (const float*)d_in[1 + 8 * blk + 5];
    const float* norm_w  = (const float*)d_in[1 + 8 * blk + 6];
    const float* out_w   = (const float*)d_in[1 + 8 * blk + 7];

    long nA = (long)NTOK * D_MODEL, nW1 = (long)D_IN_PROJ * D_MODEL;
    long nW2 = (long)D_MODEL * D_INNER;
    cvt_bf16_k<<<(nA + 255) / 256, 256, 0, stream>>>(cur, abuf, nA);
    cvt_bf16_k<<<(nW1 + 255) / 256, 256, 0, stream>>>(in_w, wbufA, nW1);
    gemm_abt<0><<<dim3((D_IN_PROJ + 127) / 128, NTOK / 128), 256, 0, stream>>>(
        abuf, wbufA, zx, NTOK, D_IN_PROJ, D_MODEL);
    conv_silu_k<<<((long)NTOK * D_CONV_CH + 255) / 256, 256, 0, stream>>>(
        zx, conv_w, conv_b, xact);
    dt_softplus_k<<<(NTOK * NHEADS + 255) / 256, 256, 0, stream>>>(zx, dt_bias, dtb);
    ssd_intra<<<dim3(NHEADS, NCHUNK, BATCH), 256, 0, stream>>>(
        xact, dtb, A_log, ybuf, states, acs);
    chunk_scan<<<dim3(NHEADS, BATCH), 256, 0, stream>>>(states, acs);
    ssd_inter<<<dim3(NHEADS, NCHUNK, BATCH), 256, 0, stream>>>(
        xact, states, acs, Dw, ybuf);
    gated_rmsnorm<<<NTOK, 256, 0, stream>>>(ybuf, zx, norm_w, ybf);
    cvt_bf16_k<<<(nW2 + 255) / 256, 256, 0, stream>>>(out_w, wbufB, nW2);
    gemm_abt<1><<<dim3(D_MODEL / 128, NTOK / 128), 256, 0, stream>>>(
        ybf, wbufB, outs[blk], NTOK, D_MODEL, D_INNER);
    cur = outs[blk];
  }
  meanpool_k<<<(BATCH * D_MODEL + 255) / 256, 256, 0, stream>>>(h2, pooled);
  heads_k<<<80, 256, 0, stream>>>(pooled, w_emo, b_emo, w_sen, b_sen, (float*)d_out);
}
